// MultiHeadCrossAttention_84542136254903
// MI455X (gfx1250) — matrix operand order repaired and checked by reading, not compiled, not run on hardware
//
#include <hip/hip_runtime.h>

typedef _Float16 v16h __attribute__((ext_vector_type(16)));
typedef _Float16 v8h  __attribute__((ext_vector_type(8)));
typedef _Float16 v4h  __attribute__((ext_vector_type(4)));
typedef float    v8f  __attribute__((ext_vector_type(8)));
typedef float    v4f  __attribute__((ext_vector_type(4)));
typedef int      v4i  __attribute__((__vector_size__(4 * sizeof(int))));

constexpr int   Bb    = 8;
constexpr int   Qq    = 1024;
constexpr int   Tt    = 1024;
constexpr int   DM    = 1024;
constexpr int   H     = 16;
constexpr int   HD    = 64;
constexpr int   WINH  = 256;       // WIN/2
constexpr float SCL   = 0.125f;    // HD^-0.5
constexpr float NEGF  = -1e30f;

#define GLOBAL_AS __attribute__((address_space(1)))
#define LDS_AS    __attribute__((address_space(3)))

#if defined(__has_builtin)
#if __has_builtin(__builtin_amdgcn_global_load_async_to_lds_b128)
#define USE_ASYNC_LDS 1
#endif
#endif
#ifndef USE_ASYNC_LDS
#define USE_ASYNC_LDS 0
#endif

__device__ inline v8f wmma_f16(v16h a, v16h b, v8f c) {
  // (neg_a, A, neg_b, B, c_mod, C, reuse_a, reuse_b)
  return __builtin_amdgcn_wmma_f32_16x16x32_f16(false, a, false, b, (short)0, c,
                                                false, false);
}

__device__ inline v8f v8f_zero() {
  v8f z = {0.f, 0.f, 0.f, 0.f, 0.f, 0.f, 0.f, 0.f};
  return z;
}

#if USE_ASYNC_LDS
// generic->AS casts via integer (LDS offset = low 32 bits of flat shared addr)
__device__ inline GLOBAL_AS v4i* as_global_v4i(const void* p) {
  return (GLOBAL_AS v4i*)(unsigned long long)p;
}
__device__ inline LDS_AS v4i* as_lds_v4i(void* p) {
  return (LDS_AS v4i*)(unsigned)(unsigned long long)p;
}
template <int N>
__device__ inline void wait_async() {
#if __has_builtin(__builtin_amdgcn_s_wait_asynccnt)
  __builtin_amdgcn_s_wait_asynccnt(N);
#else
  asm volatile("s_wait_asynccnt %0" ::"i"(N) : "memory");
#endif
}
#endif

// ---------------------------------------------------------------------------
// Elementwise f32 -> f16 (4 elems/thread, b128 in / b64 out)
// ---------------------------------------------------------------------------
__global__ __launch_bounds__(256) void cvt4_kernel(const float* __restrict__ s,
                                                   _Float16* __restrict__ d,
                                                   int n) {
  const int i = (blockIdx.x * 256 + threadIdx.x) * 4;
  if (i < n) {
    v4f f = *(const v4f*)(s + i);
    v4h h;
#pragma unroll
    for (int j = 0; j < 4; ++j) h[j] = (_Float16)f[j];
    *(v4h*)(d + i) = h;
  }
}

// ---------------------------------------------------------------------------
// C[M,N] = A[M,K] @ W[N,K]^T + bias[N]     (N == DM == 1024, A/W are f16)
// MODE 0: out = f16 row-major [M,DM]
// MODE 1: out = f32 row-major [M,DM]
// MODE 2: out = f16 transposed value layout vT[((b*H+h)*HD+d)*T + t]
// 128 threads (4 waves); block tile 64x64; K step 32.
// Staging: double-buffered global_load_async_to_lds_b128; async loads retire
// in order, so s_wait_asynccnt(4) drains tile k while tile k+1 streams in.
// ---------------------------------------------------------------------------
template <int MODE>
__global__ __launch_bounds__(128) void gemm_bias_kernel(
    const _Float16* __restrict__ A, const _Float16* __restrict__ W,
    const float* __restrict__ bias, void* __restrict__ out, int M, int K) {
  __shared__ __align__(32) _Float16 sA[2][64][40];  // 80B row stride
  __shared__ __align__(32) _Float16 sB[2][64][40];

  const int tid  = threadIdx.x;
  const int lane = tid & 31;
  const int wv   = tid >> 5;
  const int ln   = lane & 15;
  const int hf   = lane >> 4;
  const int mBase = blockIdx.y * 64;
  const int nBase = blockIdx.x * 64;

  v8f acc[4] = {v8f_zero(), v8f_zero(), v8f_zero(), v8f_zero()};

  // each K-step: 256 chunks of 8 halfs per matrix, 4 async instrs per wave
  auto compute = [&](int bufi) {
    // A fragment: lane holds row 16w+ln, halfs k = 16h..16h+15
    union { v16h v; v8h h2[2]; } af;
    const int mr = wv * 16 + ln;
    af.h2[0] = *(const v8h*)&sA[bufi][mr][16 * hf];
    af.h2[1] = *(const v8h*)&sA[bufi][mr][16 * hf + 8];
#pragma unroll
    for (int t = 0; t < 4; ++t) {
      // B fragment (W^T): lane holds col n = 16t+ln, k = 16h..16h+15
      union { v16h v; v8h h2[2]; } bf;
      const int nr = t * 16 + ln;
      bf.h2[0] = *(const v8h*)&sB[bufi][nr][16 * hf];
      bf.h2[1] = *(const v8h*)&sB[bufi][nr][16 * hf + 8];
      acc[t] = wmma_f16(af.v, bf.v, acc[t]);
    }
  };

#if USE_ASYNC_LDS
  auto issue = [&](int bufi, int kbase) {
#pragma unroll
    for (int i = 0; i < 2; ++i) {
      const int ch = i * 128 + tid;
      const int r = ch >> 2;
      const int c = (ch & 3) * 8;
      __builtin_amdgcn_global_load_async_to_lds_b128(
          as_global_v4i(A + (size_t)(mBase + r) * K + kbase + c),
          as_lds_v4i(&sA[bufi][r][c]), 0, 0);
      __builtin_amdgcn_global_load_async_to_lds_b128(
          as_global_v4i(W + (size_t)(nBase + r) * K + kbase + c),
          as_lds_v4i(&sB[bufi][r][c]), 0, 0);
    }
  };

  issue(0, 0);
  int buf = 0;
  for (int kb = 0; kb < K; kb += 32) {
    if (kb + 32 < K) {
      issue(buf ^ 1, kb + 32);  // prefetch next tile into other buffer
      wait_async<4>();          // drain current tile only (in-order retire)
    } else {
      wait_async<0>();
    }
    __syncthreads();
    compute(buf);
    __syncthreads();  // all waves done reading buf before it is re-issued
    buf ^= 1;
  }
#else
  for (int kb = 0; kb < K; kb += 32) {
#pragma unroll
    for (int i = 0; i < 2; ++i) {
      const int ch = i * 128 + tid;
      const int r = ch >> 2;
      const int c = (ch & 3) * 8;
      *(v8h*)&sA[0][r][c] = *(const v8h*)(A + (size_t)(mBase + r) * K + kb + c);
      *(v8h*)&sB[0][r][c] = *(const v8h*)(W + (size_t)(nBase + r) * K + kb + c);
    }
    __syncthreads();
    compute(0);
    __syncthreads();
  }
#endif

  // ---- epilogue: element (m = mBase+16w+8h+r, n = nBase+16t+ln) ----
#pragma unroll
  for (int t = 0; t < 4; ++t) {
#pragma unroll
    for (int r = 0; r < 8; ++r) {
      const int mrow = mBase + wv * 16 + hf * 8 + r;
      const int ncol = nBase + t * 16 + ln;
      const float v = acc[t][r] + bias[ncol];
      if (MODE == 0) {
        ((_Float16*)out)[(size_t)mrow * DM + ncol] = (_Float16)v;
      } else if (MODE == 1) {
        ((float*)out)[(size_t)mrow * DM + ncol] = v;
      } else {
        const int b = mrow >> 10, tt = mrow & (Tt - 1);  // Tt = 1024
        const int hh = ncol >> 6, d = ncol & (HD - 1);   // HD = 64
        ((_Float16*)out)[((size_t)((b * H + hh) * HD + d)) * Tt + tt] =
            (_Float16)v;
      }
    }
  }
}

// ---------------------------------------------------------------------------
// Flash-style windowed causal attention: one wave per (b, h, 16-query tile).
// Valid keys for row qi: [qi-256, qi].  Per 32-key step: 4 score WMMAs,
// online softmax (row max via 16-lane shuffles; row SUM via an extra WMMA
// against an all-ones B matrix -> normalizer lands in the correct lane slot),
// then 4 P@V WMMAs against the d-major vT layout.
// ---------------------------------------------------------------------------
__global__ __launch_bounds__(32) void attn_kernel(
    const _Float16* __restrict__ qf, const _Float16* __restrict__ kf,
    const _Float16* __restrict__ vT, const unsigned char* __restrict__ kpm,
    _Float16* __restrict__ ctx) {
  __shared__ __align__(32) _Float16 sP[16][32];  // P tile (16 q x 32 keys)

  const int lane = threadIdx.x;
  const int ln = lane & 15;
  const int hf = lane >> 4;
  const int q0 = blockIdx.x * 16;
  const int h  = blockIdx.y;
  const int b  = blockIdx.z;

  union V16 { v16h v; v8h h2[2]; };

  v16h ones;
#pragma unroll
  for (int i = 0; i < 16; ++i) ones[i] = (_Float16)1.0f;

  // q fragments: A-matrix 16x64 as two 16x32 chunks (row m = ln)
  const _Float16* qbase = qf + ((size_t)(b * Qq + q0 + ln)) * DM + h * HD;
  V16 qa[2];
#pragma unroll
  for (int c = 0; c < 2; ++c) {
    qa[c].h2[0] = *(const v8h*)(qbase + 32 * c + 16 * hf);
    qa[c].h2[1] = *(const v8h*)(qbase + 32 * c + 16 * hf + 8);
  }

  v8f o[4] = {v8f_zero(), v8f_zero(), v8f_zero(), v8f_zero()};
  v8f osum = v8f_zero();  // row sums of P (same value in every lane column)
  float mrow[8];
#pragma unroll
  for (int r = 0; r < 8; ++r) mrow[r] = NEGF;

  int start = q0 - WINH;
  if (start < 0) start = 0;
  const int end = q0 + 16;  // exclusive

  for (int kb = start; kb < end; kb += 32) {
    v8f s0 = v8f_zero(), s1 = v8f_zero();
    const int key0 = kb + ln;
    const int key1 = kb + 16 + ln;
    {
      // K^T fragments: lane = key column, 16 contiguous head-dim halfs
      const int kc0 = key0 < Tt ? key0 : Tt - 1;
      const int kc1 = key1 < Tt ? key1 : Tt - 1;
      const _Float16* kp0 = kf + (size_t)(b * Tt + kc0) * DM + h * HD + 16 * hf;
      const _Float16* kp1 = kf + (size_t)(b * Tt + kc1) * DM + h * HD + 16 * hf;
#pragma unroll
      for (int c = 0; c < 2; ++c) {
        v16h f0 = *(const v16h*)(kp0 + 32 * c);
        v16h f1 = *(const v16h*)(kp1 + 32 * c);
        s0 = wmma_f16(qa[c].v, f0, s0);
        s1 = wmma_f16(qa[c].v, f1, s1);
      }
    }

    // ---- mask + scale, online softmax (rows = 8h+r across 16-lane half) ----
    const bool pad0 = (key0 < Tt) ? (kpm[b * Tt + key0] == 0) : true;
    const bool pad1 = (key1 < Tt) ? (kpm[b * Tt + key1] == 0) : true;
    float mt[8];
#pragma unroll
    for (int r = 0; r < 8; ++r) {
      const int qi = q0 + 8 * hf + r;
      const bool m0 = (key0 > qi) || (key0 < qi - WINH) || pad0;
      const bool m1 = (key1 > qi) || (key1 < qi - WINH) || pad1;
      s0[r] = m0 ? NEGF : s0[r] * SCL;
      s1[r] = m1 ? NEGF : s1[r] * SCL;
      mt[r] = fmaxf(s0[r], s1[r]);
    }
#pragma unroll
    for (int off = 1; off < 16; off <<= 1)
#pragma unroll
      for (int r = 0; r < 8; ++r)
        mt[r] = fmaxf(mt[r], __shfl_xor(mt[r], off, 32));

    float alpha[8];
#pragma unroll
    for (int r = 0; r < 8; ++r) {
      const float mn = fmaxf(mrow[r], mt[r]);
      alpha[r] = __expf(mrow[r] - mn);
      const float p0 = __expf(s0[r] - mn);
      const float p1 = __expf(s1[r] - mn);
      sP[8 * hf + r][ln] = (_Float16)p0;
      sP[8 * hf + r][16 + ln] = (_Float16)p1;
      mrow[r] = mn;
    }
    // rescale running accumulators (incl. normalizer) by alpha
#pragma unroll
    for (int r = 0; r < 8; ++r) osum[r] *= alpha[r];
#pragma unroll
    for (int t = 0; t < 4; ++t)
#pragma unroll
      for (int r = 0; r < 8; ++r) o[t][r] *= alpha[r];

    __syncthreads();  // single-wave WG: ensures P visible in A layout

    V16 pa;
    pa.h2[0] = *(const v8h*)&sP[ln][16 * hf];
    pa.h2[1] = *(const v8h*)&sP[ln][16 * hf + 8];
    // normalizer: osum += P @ ones  (row sum broadcast to every column slot)
    osum = wmma_f16(pa.v, ones, osum);
#pragma unroll
    for (int t = 0; t < 4; ++t) {
      // V B-fragment: col n = d = 16t+ln, 16 contiguous keys at kb+16h
      const _Float16* vb =
          vT + ((size_t)((b * H + h) * HD + t * 16 + ln)) * Tt + kb + 16 * hf;
      v16h vfr = *(const v16h*)vb;
      o[t] = wmma_f16(pa.v, vfr, o[t]);
    }
    __syncthreads();  // protect sP WAR across iterations
  }

  float inv[8];
#pragma unroll
  for (int r = 0; r < 8; ++r) inv[r] = 1.0f / osum[r];
#pragma unroll
  for (int t = 0; t < 4; ++t)
#pragma unroll
    for (int r = 0; r < 8; ++r) {
      const int qi = q0 + 8 * hf + r;
      const int d = t * 16 + ln;
      ctx[((size_t)(b * Qq + qi)) * DM + h * HD + d] =
          (_Float16)(o[t][r] * inv[r]);
    }
}

// ---------------------------------------------------------------------------
extern "C" void kernel_launch(void* const* d_in, const int* in_sizes, int n_in,
                              void* d_out, int out_size, void* d_ws,
                              size_t ws_size, hipStream_t stream) {
  (void)in_sizes; (void)n_in; (void)out_size; (void)ws_size;
  const float* query = (const float*)d_in[0];
  const float* key_  = (const float*)d_in[1];
  const float* value = (const float*)d_in[2];
  const float* Wq = (const float*)d_in[3];
  const float* bq = (const float*)d_in[4];
  const float* Wk = (const float*)d_in[5];
  const float* bk = (const float*)d_in[6];
  const float* Wv = (const float*)d_in[7];
  const float* bv = (const float*)d_in[8];
  const float* Wo = (const float*)d_in[9];
  const float* bo = (const float*)d_in[10];
  const unsigned char* kpm = (const unsigned char*)d_in[11];  // jax bool = 1B
  float* out = (float*)d_out;

  const size_t MT = (size_t)Bb * Qq;  // 8192 rows (== B*T)

  // workspace layout (all _Float16)
  _Float16* w = (_Float16*)d_ws;
  _Float16* qh  = w;                 w += MT * 128;   // query f16
  _Float16* kh  = w;                 w += MT * 256;   // key f16
  _Float16* vh  = w;                 w += MT * 256;   // value f16
  _Float16* Wqh = w;                 w += (size_t)DM * 128;
  _Float16* Wkh = w;                 w += (size_t)DM * 256;
  _Float16* Wvh = w;                 w += (size_t)DM * 256;
  _Float16* Woh = w;                 w += (size_t)DM * DM;
  _Float16* qf  = w;                 w += MT * DM;    // projected q
  _Float16* kf  = w;                 w += MT * DM;    // projected k
  _Float16* vT  = w;                 w += MT * DM;    // projected v, d-major
  _Float16* ctx = w;                 w += MT * DM;    // attention output

  // ---- pre-convert everything the WMMA GEMMs consume to f16 ----
  auto cvt = [&](const float* s, _Float16* d, size_t n) {
    hipLaunchKernelGGL(cvt4_kernel, dim3((unsigned)(n / 1024)), dim3(256), 0,
                       stream, s, d, (int)n);
  };
  cvt(query, qh, MT * 128);
  cvt(key_,  kh, MT * 256);
  cvt(value, vh, MT * 256);
  cvt(Wq, Wqh, (size_t)DM * 128);
  cvt(Wk, Wkh, (size_t)DM * 256);
  cvt(Wv, Wvh, (size_t)DM * 256);
  cvt(Wo, Woh, (size_t)DM * DM);

  const dim3 blk(128);
  const dim3 gP(DM / 64, (unsigned)(MT / 64));  // (16, 128)

  hipLaunchKernelGGL((gemm_bias_kernel<0>), gP, blk, 0, stream,
                     qh, Wqh, bq, (void*)qf, (int)MT, 128);
  hipLaunchKernelGGL((gemm_bias_kernel<0>), gP, blk, 0, stream,
                     kh, Wkh, bk, (void*)kf, (int)MT, 256);
  hipLaunchKernelGGL((gemm_bias_kernel<2>), gP, blk, 0, stream,
                     vh, Wvh, bv, (void*)vT, (int)MT, 256);
  hipLaunchKernelGGL(attn_kernel, dim3(Qq / 16, H, Bb), dim3(32), 0, stream,
                     qf, kf, vT, kpm, ctx);
  hipLaunchKernelGGL((gemm_bias_kernel<1>), gP, blk, 0, stream,
                     ctx, Woh, bo, (void*)out, (int)MT, 1024);
}
